// PyramidAttentionModule_66408784331550
// MI455X (gfx1250) — matrix operand order repaired and checked by reading, not compiled, not run on hardware
//
#include <hip/hip_runtime.h>
#include <hip/hip_bf16.h>

typedef __bf16 bf16_t;
typedef __attribute__((ext_vector_type(16))) __bf16 v16bf;
typedef __attribute__((ext_vector_type(8)))  __bf16 v8bf;
typedef __attribute__((ext_vector_type(8)))  float   v8f;
typedef __attribute__((ext_vector_type(4)))  float   v4f;

#define WMMA_BF16(A,B,C) __builtin_amdgcn_wmma_f32_16x16x32_bf16(false,(A),false,(B),(short)0,(C),false,false)

// problem dims
#define BB    2
#define CIN   256
#define HH    64
#define WW2   128
#define KCQ   32
#define VCC   256
#define HWSZ  8192      // HH*WW2
#define NPIX  16384     // BB*HWSZ
#define EPSB  1e-5f
#define KSCL  0.17677669529663689f  // 32^-0.5

// ---------------- workspace layout (bytes) ----------------
static constexpr size_t XT_OFF = 0;
static constexpr size_t XT_SZ  = (size_t)NPIX * CIN * 2;          // NHWC bf16 input
static constexpr size_t QSC    = (size_t)16384 * 32;              // q/k elements per scale
static constexpr size_t VSC    = (size_t)16384 * 256;             // v elements per scale
static constexpr size_t Q_OFF  = XT_OFF + XT_SZ;
static constexpr size_t K_OFF  = Q_OFF + 4 * QSC * 2;
static constexpr size_t V_OFF  = K_OFF + 4 * QSC * 2;
static constexpr size_t ST_OFF = V_OFF + 4 * VSC * 2;
static constexpr size_t ST_SC  = (size_t)16384 * 2;               // floats per scale (max, inv)
static constexpr size_t CTX_OFF= ST_OFF + 4 * ST_SC * 4;
static constexpr size_t CTX_SZ = (size_t)NPIX * 1024 * 2;         // NHWC bf16
static constexpr size_t WT_OFF = CTX_OFF + CTX_SZ;                // 9*256*1024 bf16
static constexpr size_t WT_SZ  = (size_t)9*256*1024 * 2;
static constexpr size_t WQ_OFF = WT_OFF + WT_SZ;                  // 4*32*256 bf16 (BN+KSCL folded)
static constexpr size_t WQ_SZ  = (size_t)4*32*256 * 2;
static constexpr size_t WK_OFF = WQ_OFF + WQ_SZ;                  // 4*32*256 bf16 (BN folded)
static constexpr size_t WV_OFF = WK_OFF + WQ_SZ;                  // 4*256*256 bf16
static constexpr size_t WV_SZ  = (size_t)4*256*256 * 2;
static constexpr size_t QB_OFF = WV_OFF + WV_SZ;                  // 128 f32 folded q bias
static constexpr size_t KB_OFF = QB_OFF + 512;                    // 128 f32 folded k bias

// ---------------- helpers ----------------
__device__ inline v8bf ld8(const bf16_t* p){ return *(const v8bf*)p; }
__device__ inline v16bf cat16(v8bf a, v8bf b){
  return __builtin_shufflevector(a,b,0,1,2,3,4,5,6,7,8,9,10,11,12,13,14,15);
}
// A-fragment (row-per-lane, K interleave {0-7,16-23}/{8-15,24-31})
__device__ inline v16bf afrag(const bf16_t* row, int hl){
  const int g1 = hl ? 8 : 0, g2 = hl ? 24 : 16;
  return cat16(ld8(row + g1), ld8(row + g2));
}
// B-fragment (column-per-lane, K interleave {0-7,16-23}/{8-15,24-31})
__device__ inline v16bf bfrag(const bf16_t* col, int hl){
  const int g1 = hl ? 8 : 0, g2 = hl ? 24 : 16;
  return cat16(ld8(col + g1), ld8(col + g2));
}
// window (wb,l) -> flat pixel index (b*HH + h)*WW2 + w
__device__ inline int pix_off(int sc, int wb, int l){
  int bb = wb & 1; int ij = wb >> 1;
  int j = ij & ((1<<sc)-1); int i = ij >> sc;
  int side = l & 1; int t = l >> 1;
  int wsl = 6 - sc;
  int x = t & ((1<<wsl)-1); int y = t >> wsl;
  int h = (i << wsl) + y;
  int w = side*64 + (j << wsl) + x;
  return (bb*HH + h)*WW2 + w;
}
__device__ inline void decode_wave(int wid, int& sc, int& wb, int& l0, int& L){
  sc = wid >> 10; int r = wid & 1023;
  L = HWSZ >> (2*sc);
  int ntl = 9 - 2*sc;                  // log2(query tiles per window batch)
  wb = r >> ntl; l0 = (r & ((1<<ntl)-1)) << 4;
}

// ---------------- kernel 1: NCHW fp32 -> NHWC bf16 ----------------
__global__ void k_xt(const float* __restrict__ feats, bf16_t* __restrict__ xt){
  __shared__ float tile[32][257];
  int bidx = blockIdx.x;                          // 512 blocks
  int b = bidx / (HWSZ/32); int hw0 = (bidx % (HWSZ/32)) * 32;
#pragma unroll 4
  for(int it=0; it<32; ++it){
    int flat = it*256 + threadIdx.x; int c = flat >> 5; int p = flat & 31;
    tile[p][c] = feats[((size_t)b*CIN + c)*HWSZ + hw0 + p];
  }
  __syncthreads();
#pragma unroll 4
  for(int it=0; it<32; ++it){
    int flat = it*256 + threadIdx.x; int p = flat >> 8; int c = flat & 255;
    xt[((size_t)(b*HWSZ + hw0 + p))*CIN + c] = (bf16_t)tile[p][c];
  }
}

// ---------------- kernel 2: conv3x3 weight transform (BN-scale folded) ----------------
__global__ void k_wt(const float* __restrict__ cw, const float* __restrict__ cg,
                     const float* __restrict__ cv, bf16_t* __restrict__ wt){
  int gid = blockIdx.x*256 + threadIdx.x;
  if(gid >= 9*256*1024) return;
  int tap = gid / (256*1024); int rem = gid - tap*256*1024;
  int o = rem >> 10; int ci = rem & 1023;
  float sb = cg[o]*rsqrtf(cv[o]+EPSB);
  wt[gid] = (bf16_t)(cw[((size_t)o*1024 + ci)*9 + tap] * sb);
}

// ---------------- kernel 2b: q/k/v projection weight transform ----------------
// wq[sc][o][c] = qw * bn_scale * KSCL ; wk[sc][o][c] = kw * bn_scale ; wv[sc][c][k] = vw
__global__ void k_wqkv(const float* __restrict__ qw, const float* __restrict__ qg,
                       const float* __restrict__ qv,
                       const float* __restrict__ kw, const float* __restrict__ kg,
                       const float* __restrict__ kvv,
                       const float* __restrict__ vw,
                       bf16_t* __restrict__ wq, bf16_t* __restrict__ wk,
                       bf16_t* __restrict__ wv){
  int gid = blockIdx.x*256 + threadIdx.x;          // 327680 total
  if(gid < 32768){
    int o = gid >> 8;                              // sc*32 + oc
    float sb = qg[o]*rsqrtf(qv[o]+EPSB);
    wq[gid] = (bf16_t)(qw[gid] * sb * KSCL);
  } else if(gid < 65536){
    int g = gid - 32768; int o = g >> 8;
    float sb = kg[o]*rsqrtf(kvv[o]+EPSB);
    wk[g] = (bf16_t)(kw[g] * sb);
  } else if(gid < 65536 + 262144){
    int g = gid - 65536;
    wv[g] = (bf16_t)vw[g];
  }
}

// ---------------- kernel 2c: folded q/k biases ----------------
__global__ void k_bias(const float* __restrict__ qb, const float* __restrict__ qg,
                       const float* __restrict__ qbe, const float* __restrict__ qm,
                       const float* __restrict__ qv,
                       const float* __restrict__ kb, const float* __restrict__ kg,
                       const float* __restrict__ kbe, const float* __restrict__ km,
                       const float* __restrict__ kvv,
                       float* __restrict__ qbias, float* __restrict__ kbias){
  int t = threadIdx.x;                             // 256 threads, 1 block
  if(t < 128){
    float sb = qg[t]*rsqrtf(qv[t]+EPSB);
    qbias[t] = ((qb[t]-qm[t])*sb + qbe[t]) * KSCL;
  } else {
    int o = t - 128;
    float sb = kg[o]*rsqrtf(kvv[o]+EPSB);
    kbias[o] = (kb[o]-km[o])*sb + kbe[o];
  }
}

// ---------------- kernel 3: fused q/k/v projections (per window tile) ----------------
__global__ void __launch_bounds__(128,1)
k_proj(const bf16_t* __restrict__ xt,
       const bf16_t* __restrict__ wq, const bf16_t* __restrict__ wk,
       const bf16_t* __restrict__ wv,
       const float* __restrict__ qbias, const float* __restrict__ kbias,
       const float* __restrict__ vb,
       bf16_t* __restrict__ qo, bf16_t* __restrict__ ko, bf16_t* __restrict__ vo){
  int wid = (blockIdx.x*blockDim.x + threadIdx.x) >> 5;
  int lane = threadIdx.x & 31; int hl = lane >> 4; int l16 = lane & 15;
  int sc, wb, l0, L; decode_wave(wid, sc, wb, l0, L);
  int myl = l0 + l16;
  const bf16_t* xrow = xt + (size_t)pix_off(sc, wb, myl)*CIN;

  // preload X fragments for all 8 K-steps (A-rows for v-form, B-cols for q/k-form)
  v16bf xa[8], xb[8];
#pragma unroll
  for(int ks=0; ks<8; ++ks){
    xa[ks] = afrag(xrow + ks*32, hl);
    xb[ks] = bfrag(xrow + ks*32, hl);
  }

  // ---- Q and K : D[o, l] = W[o,:] X[:,l] ; BN (+KC^-0.5 for Q) pre-folded ----
#pragma unroll
  for(int which=0; which<2; ++which){
    const bf16_t* W    = which ? wk    : wq;
    const float*  bias = which ? kbias : qbias;
    bf16_t*       dst  = which ? ko    : qo;
#pragma unroll
    for(int ot=0; ot<2; ++ot){
      const bf16_t* wrow = W + (size_t)(sc*KCQ + ot*16 + l16)*CIN;
      v8f acc = {};
#pragma unroll
      for(int ks=0; ks<8; ++ks)
        acc = WMMA_BF16(afrag(wrow + ks*32, hl), xb[ks], acc);
      v8bf pk;
#pragma unroll
      for(int rr=0; rr<8; ++rr)
        pk[rr] = (bf16_t)(acc[rr] + bias[sc*KCQ + ot*16 + hl*8 + rr]);
      *(v8bf*)(dst + (size_t)sc*QSC + ((size_t)wb*L + myl)*32 + ot*16 + hl*8) = pk;
    }
  }

  // ---- V : D[l, c] = X^T[l,:] W^T[:,c] ----
#pragma unroll
  for(int ct=0; ct<16; ++ct){
    int c = ct*16 + l16;
    const bf16_t* wrow = wv + (size_t)(sc*VCC + c)*CIN;
    v8f acc = {};
#pragma unroll
    for(int ks=0; ks<8; ++ks)
      acc = WMMA_BF16(xa[ks], bfrag(wrow + ks*32, hl), acc);
    float bias = vb[sc*VCC + c];
    v8bf pk;
#pragma unroll
    for(int rr=0; rr<8; ++rr) pk[rr] = (bf16_t)(acc[rr] + bias);
    *(v8bf*)(vo + (size_t)sc*VSC + ((size_t)wb*VCC + c)*L + l0 + hl*8) = pk;
  }
}

// ---------------- kernel 4: softmax stats (row max + 1/sum) ----------------
__global__ void __launch_bounds__(128,1)
k_stats(const bf16_t* __restrict__ qo, const bf16_t* __restrict__ ko,
        float* __restrict__ stats){
  int wid = (blockIdx.x*blockDim.x + threadIdx.x) >> 5;
  int lane = threadIdx.x & 31; int hl = lane >> 4; int l16 = lane & 15;
  int sc, wb, l0, L; decode_wave(wid, sc, wb, l0, L);
  int myl = l0 + l16;
  v16bf qf = bfrag(qo + (size_t)sc*QSC + ((size_t)wb*L + myl)*32, hl);
  const bf16_t* kb2 = ko + (size_t)sc*QSC + (size_t)wb*L*32;

  float m = -3.0e38f, Z = 0.f;
  for(int m0=0; m0<L; m0+=16){
    if(m0 + 16 < L) __builtin_prefetch(kb2 + (size_t)(m0 + 16 + l16)*32, 0, 1);
    v16bf kf = afrag(kb2 + (size_t)(m0 + l16)*32, hl);
    v8f s = {};
    s = WMMA_BF16(kf, qf, s);
    float tm = s[0];
#pragma unroll
    for(int i=1;i<8;i++) tm = fmaxf(tm, s[i]);
    float mn = fmaxf(m, tm);
    float zz = Z * __expf(m - mn);
#pragma unroll
    for(int i=0;i<8;i++) zz += __expf(s[i] - mn);
    m = mn; Z = zz;
  }
  // combine lane l (keys 16t+0..7) with lane l+16 (keys 16t+8..15)
  float mo = __shfl_xor(m, 16);
  float Zo = __shfl_xor(Z, 16);
  float mn = fmaxf(m, mo);
  float Zt = Z*__expf(m - mn) + Zo*__expf(mo - mn);
  if(hl == 0){
    float* st = stats + (size_t)sc*ST_SC;
    int qi = wb*L + l0 + l16;
    st[qi] = mn;
    st[16384 + qi] = 1.0f / Zt;
  }
}

// ---------------- kernel 5: attention P.V (flash-style, stats precomputed) ----------------
__global__ void __launch_bounds__(128,1)
k_attn(const bf16_t* __restrict__ qo, const bf16_t* __restrict__ ko,
       const bf16_t* __restrict__ vo, const float* __restrict__ stats,
       bf16_t* __restrict__ ctx){
  __shared__ __align__(16) bf16_t plds[4][16][40];   // 80B row stride: conflict-free b128
  int wIb = threadIdx.x >> 5;
  int wid = (blockIdx.x*blockDim.x + threadIdx.x) >> 5;
  int lane = threadIdx.x & 31; int hl = lane >> 4; int l16 = lane & 15;
  int sc, wb, l0, L; decode_wave(wid, sc, wb, l0, L);
  int myl = l0 + l16;

  v16bf qf = bfrag(qo + (size_t)sc*QSC + ((size_t)wb*L + myl)*32, hl);
  const float* st = stats + (size_t)sc*ST_SC;
  float mx = st[wb*L + myl];
  float iz = st[16384 + wb*L + myl];

  const bf16_t* kb2   = ko + (size_t)sc*QSC + (size_t)wb*L*32;
  const bf16_t* vbase = vo + (size_t)sc*VSC + (size_t)wb*VCC*L;
  bf16_t (*P)[40] = plds[wIb];

  v8f acc[16] = {};
  for(int m0=0; m0<L; m0+=32){
    if(m0 + 32 < L) __builtin_prefetch(kb2 + (size_t)(m0 + 32 + l16)*32, 0, 1);
#pragma unroll
    for(int t=0; t<2; ++t){
      v16bf kf = afrag(kb2 + (size_t)(m0 + t*16 + l16)*32, hl);
      v8f s = {};
      s = WMMA_BF16(kf, qf, s);                     // D[m, l]: lane holds query col l16
      v8bf pk;
#pragma unroll
      for(int rr=0; rr<8; ++rr) pk[rr] = (bf16_t)(__expf(s[rr] - mx) * iz);
      *(v8bf*)&P[l16][t*16 + hl*8] = pk;            // P[q][m] row-major in m
    }
    asm volatile("s_wait_dscnt 0" ::: "memory");
    v16bf pf = bfrag(&P[l16][0], hl);               // B-frag: col q, keys {0-7,16-23}/{8-15,24-31}
#pragma unroll
    for(int ct=0; ct<16; ++ct){
      const bf16_t* vrow = vbase + (size_t)(ct*16 + l16)*L + m0;
      acc[ct] = WMMA_BF16(afrag(vrow, hl), pf, acc[ct]);   // D[c, l]
    }
  }
  // store ctx NHWC: lane holds query col -> pixel; rows are 8 consecutive channels
  bf16_t* crow = ctx + (size_t)pix_off(sc, wb, myl)*1024 + sc*VCC;
#pragma unroll
  for(int ct=0; ct<16; ++ct){
    v8bf pk;
#pragma unroll
    for(int rr=0; rr<8; ++rr) pk[rr] = (bf16_t)acc[ct][rr];
    *(v8bf*)(crow + ct*16 + hl*8) = pk;
  }
}

// ---------------- kernel 6: 3x3 conv + bias + BN + ReLU ----------------
__global__ void __launch_bounds__(128,1)
k_conv(const bf16_t* __restrict__ ctx, const bf16_t* __restrict__ wt,
       const float* __restrict__ cb, const float* __restrict__ cg,
       const float* __restrict__ cbe, const float* __restrict__ cm,
       const float* __restrict__ cv, float* __restrict__ out){
  int wid = (blockIdx.x*blockDim.x + threadIdx.x) >> 5;   // 0..1023
  int lane = threadIdx.x & 31; int hl = lane >> 4; int l16 = lane & 15;
  int b = wid >> 9; int h = (wid >> 3) & 63; int w0 = (wid & 7) << 4;

  v16bf zf;
#pragma unroll
  for(int i=0;i<16;i++) zf[i] = (bf16_t)0.0f;

  v8f acc[16] = {};
#pragma unroll
  for(int dh=-1; dh<=1; ++dh){
    int hh = h + dh;
    if(hh < 0 || hh >= HH) continue;
#pragma unroll
    for(int dw=-1; dw<=1; ++dw){
      int ww = w0 + l16 + dw;
      bool val = (ww >= 0 && ww < WW2);
      const bf16_t* arow = ctx + (size_t)((b*HH + hh)*WW2 + (val ? ww : 0))*1024;
      const bf16_t* wtap = wt + (size_t)((dh+1)*3 + (dw+1))*256*1024;
      for(int ks=0; ks<32; ++ks){
        v16bf af = val ? afrag(arow + ks*32, hl) : zf;
#pragma unroll
        for(int ot=0; ot<16; ++ot){
          const bf16_t* wcol = wtap + (size_t)(ot*16 + l16)*1024 + ks*32;
          acc[ot] = WMMA_BF16(af, bfrag(wcol, hl), acc[ot]);  // D[pixel, o]
        }
      }
    }
  }
  // epilogue: weights already carry BN scale; add folded bias/shift, ReLU, store fp32 NCHW
#pragma unroll
  for(int ot=0; ot<16; ++ot){
    int o = ot*16 + l16;
    float sb = cg[o]*rsqrtf(cv[o]+EPSB);
    float beta = cb[o]*sb + cbe[o] - cm[o]*sb;
    v4f r0, r1;
#pragma unroll
    for(int rr=0; rr<8; ++rr){
      float v = fmaxf(acc[ot][rr] + beta, 0.f);
      if(rr < 4) r0[rr] = v; else r1[rr-4] = v;
    }
    float* orow = out + ((size_t)(b*256 + o)*HH + h)*WW2 + w0 + hl*8;
    *(v4f*)orow = r0;
    *(v4f*)(orow + 4) = r1;
  }
}

// ---------------- launcher ----------------
extern "C" void kernel_launch(void* const* d_in, const int* in_sizes, int n_in,
                              void* d_out, int out_size, void* d_ws, size_t ws_size,
                              hipStream_t stream) {
  const float* feats = (const float*)d_in[0];
  const float* q_w = (const float*)d_in[1];  const float* q_b = (const float*)d_in[2];
  const float* q_g = (const float*)d_in[3];  const float* q_be = (const float*)d_in[4];
  const float* q_m = (const float*)d_in[5];  const float* q_v = (const float*)d_in[6];
  const float* k_w = (const float*)d_in[7];  const float* k_b = (const float*)d_in[8];
  const float* k_g = (const float*)d_in[9];  const float* k_be = (const float*)d_in[10];
  const float* k_m = (const float*)d_in[11]; const float* k_v = (const float*)d_in[12];
  const float* v_w = (const float*)d_in[13]; const float* v_b = (const float*)d_in[14];
  const float* conv_w = (const float*)d_in[15]; const float* conv_b = (const float*)d_in[16];
  const float* c_g = (const float*)d_in[17]; const float* c_be = (const float*)d_in[18];
  const float* c_m = (const float*)d_in[19]; const float* c_v = (const float*)d_in[20];

  char* ws = (char*)d_ws;
  bf16_t* xt  = (bf16_t*)(ws + XT_OFF);
  bf16_t* qo  = (bf16_t*)(ws + Q_OFF);
  bf16_t* ko  = (bf16_t*)(ws + K_OFF);
  bf16_t* vo  = (bf16_t*)(ws + V_OFF);
  float*  stt = (float*)(ws + ST_OFF);
  bf16_t* ctx = (bf16_t*)(ws + CTX_OFF);
  bf16_t* wtb = (bf16_t*)(ws + WT_OFF);
  bf16_t* wqb = (bf16_t*)(ws + WQ_OFF);
  bf16_t* wkb = (bf16_t*)(ws + WK_OFF);
  bf16_t* wvb = (bf16_t*)(ws + WV_OFF);
  float*  qbb = (float*)(ws + QB_OFF);
  float*  kbb = (float*)(ws + KB_OFF);

  k_xt  <<<NPIX/32, 256, 0, stream>>>(feats, xt);
  k_wt  <<<(9*256*1024)/256, 256, 0, stream>>>(conv_w, c_g, c_v, wtb);
  k_wqkv<<<(327680+255)/256, 256, 0, stream>>>(q_w, q_g, q_v, k_w, k_g, k_v, v_w,
                                               wqb, wkb, wvb);
  k_bias<<<1, 256, 0, stream>>>(q_b, q_g, q_be, q_m, q_v,
                                k_b, k_g, k_be, k_m, k_v, qbb, kbb);
  k_proj<<<1024, 128, 0, stream>>>(xt, wqb, wkb, wvb, qbb, kbb, v_b, qo, ko, vo);
  k_stats<<<1024, 128, 0, stream>>>(qo, ko, stt);
  k_attn <<<1024, 128, 0, stream>>>(qo, ko, vo, stt, ctx);
  k_conv <<<256, 128, 0, stream>>>(ctx, wtb, conv_b, c_g, c_be, c_m, c_v, (float*)d_out);
}
